// SubGDiscriminator_5944234737771
// MI455X (gfx1250) — matrix operand order repaired and checked by reading, not compiled, not run on hardware
//
#include <hip/hip_runtime.h>
#include <hip/hip_bf16.h>

// ---------------------------------------------------------------------------
// Problem constants (from the reference file)
// ---------------------------------------------------------------------------
#define GN2 2000
#define GN1 20000
#define GN0 50000
#define GH  256

typedef __attribute__((ext_vector_type(16))) __bf16 v16bf;
typedef __attribute__((ext_vector_type(8)))  float  v8f;

// ---------------------------------------------------------------------------
// WMMA 16x16x32 bf16 operand layouts, per CDNA5 ISA 7.12.2:
// A (16x32, MxK), lane L: M = L%16; half i holds K = (i/8)*16 + (L>=16)*8 + i%8.
// B (32x16, KxN), lane L: N = L%16; half i holds K = (i/8)*16 + (L>=16)*8 + i%8.
// C/D (16x16 f32), lane L, vgpr r: M = r + 8*(L>=16), N = L%16.
// ---------------------------------------------------------------------------
__device__ __forceinline__ v16bf load_a16x32(const float* __restrict__ X,
                                             int row0, int kb, int lane) {
  const int m  = row0 + (lane & 15);
  const int hi = (lane >> 4) & 1;
  const float* xr = X + (size_t)m * GH + kb;
  v16bf a;
#pragma unroll
  for (int i = 0; i < 16; ++i) {
    const int k = ((i >> 3) << 4) + (hi << 3) + (i & 7);
    a[i] = (__bf16)xr[k];
  }
  return a;
}

// ---------------------------------------------------------------------------
// One-shot W pre-pack: f32 (256x256 row-major) -> bf16, swizzled into the
// exact per-lane B-operand order. Group g = colTile*8 + kBlock; lane L's 16
// halves land contiguously so the GEMM fetches a whole B tile as one 32-byte
// vector load (2x global_load_b128, zero cvt in the hot loop). 128 KB/half,
// trivially L2-resident on MI455X (192 MB L2).
// ---------------------------------------------------------------------------
__global__ void pack_w_bf16(const float* __restrict__ W, __bf16* __restrict__ Wp) {
  const int lane = threadIdx.x & 31;
  const int g    = blockIdx.x * (blockDim.x >> 5) + (threadIdx.x >> 5); // 0..127
  const int ct   = g >> 3;          // column tile 0..15
  const int kb   = (g & 7) << 5;    // K block base
  const int n    = (ct << 4) + (lane & 15);
  const int hi   = (lane >> 4) & 1;
  v16bf b;
#pragma unroll
  for (int i = 0; i < 16; ++i) {
    const int k = kb + ((i >> 3) << 4) + (hi << 3) + (i & 7);
    b[i] = (__bf16)W[(size_t)k * GH + n];
  }
  *(v16bf*)(Wp + ((size_t)g * 32 + lane) * 16) = b;
}

// ---------------------------------------------------------------------------
// Y[M x 256] = X[M x 256] @ W[256 x 256]  (bf16 WMMA, f32 accum)
// blockDim = 256 (8 wave32 waves). Block b owns rows [16b,16b+16); all 8 A
// K-tiles are loaded/converted once into registers, then each wave computes
// two 16x16 column tiles: inner loop = packed-B v16bf load + v_wmma.
// M is always a multiple of 16 here.
// ---------------------------------------------------------------------------
__global__ void proj_gemm_bf16(const float* __restrict__ X,
                               const __bf16* __restrict__ Wp,
                               float* __restrict__ Y, int M) {
  const int lane = threadIdx.x & 31;
  const int wave = threadIdx.x >> 5;
  const int row0 = blockIdx.x << 4;
  if (row0 >= M) return;

  v16bf a[8];
#pragma unroll
  for (int kk = 0; kk < 8; ++kk) a[kk] = load_a16x32(X, row0, kk << 5, lane);

#pragma unroll
  for (int t = 0; t < 2; ++t) {
    const int ct = (wave << 1) + t;   // column tile 0..15
    v8f acc = {};
#pragma unroll
    for (int kk = 0; kk < 8; ++kk) {
      const v16bf b =
          *(const v16bf*)(Wp + (((size_t)ct * 8 + kk) * 32 + lane) * 16);
      acc = __builtin_amdgcn_wmma_f32_16x16x32_bf16(
          /*neg_a=*/false, a[kk], /*neg_b=*/false, b,
          /*c_mod=*/(short)0, acc, /*reuse_a=*/false, /*reuse_b=*/false);
    }
    const int hi = (lane >> 4) & 1;
    const int n  = (ct << 4) + (lane & 15);
#pragma unroll
    for (int r = 0; r < 8; ++r)
      Y[(size_t)(row0 + r + (hi << 3)) * GH + n] = acc[r];
  }
}

// ---------------------------------------------------------------------------
// Workspace zeroing (Asum + cnt must be 0 before the scatter pass; harness
// poisons d_ws once and never re-poisons between replays).
// ---------------------------------------------------------------------------
__global__ void zero_kernel(float* __restrict__ p, int n) {
  int i = blockIdx.x * blockDim.x + threadIdx.x;
  const int stride = gridDim.x * blockDim.x;
  for (; i < n; i += stride) p[i] = 0.0f;
}

// ---------------------------------------------------------------------------
// Segment-sum scatter: Asum[dst[e]] += A[src[e]], cnt[dst[e]] += 1.
// 64 lanes (4 floats each) per edge, 4 edges per 256-thread block. The 2 MB
// A array and 20 MB Asum are L2-resident; f32 atomics resolve in L2.
// ---------------------------------------------------------------------------
__global__ void scatter_add_kernel(const float* __restrict__ A,
                                   const int* __restrict__ src,
                                   const int* __restrict__ dst,
                                   float* __restrict__ Asum,
                                   float* __restrict__ cnt, int E) {
  const int e = blockIdx.x * (blockDim.x >> 6) + (threadIdx.x >> 6);
  if (e >= E) return;
  const int t = threadIdx.x & 63;
  const int s = src[e];
  const int d = dst[e];
  const float4 v = *(const float4*)(A + (size_t)s * GH + (t << 2));
  float* o = Asum + (size_t)d * GH + (t << 2);
  atomicAdd(o + 0, v.x);
  atomicAdd(o + 1, v.y);
  atomicAdd(o + 2, v.z);
  atomicAdd(o + 3, v.w);
  if (t == 0) atomicAdd(cnt + d, 1.0f);
}

// A0 = Asum / max(cnt, 1)  (in place; cnt==0 rows stay 0, matching has_msg)
__global__ void scale_kernel(float* __restrict__ Asum,
                             const float* __restrict__ cnt) {
  const int d = blockIdx.x;
  const float inv = 1.0f / fmaxf(cnt[d], 1.0f);
  Asum[(size_t)d * GH + threadIdx.x] *= inv;
}

// ---------------------------------------------------------------------------
// Edge stage: out[e] = relu(Asrc[src[e]] + Bdst[dst[e]] + b_lin) . W_us + b_us
// One wave32 per edge, 8 floats per lane, shuffle reduction. Gather operands
// (<= 92 MB total) are L2-resident on MI455X.
// ---------------------------------------------------------------------------
__global__ void edge_out_kernel(const float* __restrict__ Asrc,
                                const float* __restrict__ Bdst,
                                const int* __restrict__ src,
                                const int* __restrict__ dst,
                                const float* __restrict__ b_lin,
                                const float* __restrict__ W_us,
                                const float* __restrict__ b_us,
                                float* __restrict__ out, int E) {
  const int e = blockIdx.x * (blockDim.x >> 5) + (threadIdx.x >> 5);
  if (e >= E) return;
  const int lane = threadIdx.x & 31;
  const int s = src[e];
  const int d = dst[e];
  const float* pa = Asrc + (size_t)s * GH;
  const float* pb = Bdst + (size_t)d * GH;

  float acc = 0.0f;
#pragma unroll
  for (int h = 0; h < 2; ++h) {
    const int off = (lane << 3) + (h << 2);
    const float4 a  = *(const float4*)(pa + off);
    const float4 b  = *(const float4*)(pb + off);
    const float4 bl = *(const float4*)(b_lin + off);
    const float4 w  = *(const float4*)(W_us + off);
    acc += fmaxf(a.x + b.x + bl.x, 0.0f) * w.x;
    acc += fmaxf(a.y + b.y + bl.y, 0.0f) * w.y;
    acc += fmaxf(a.z + b.z + bl.z, 0.0f) * w.z;
    acc += fmaxf(a.w + b.w + bl.w, 0.0f) * w.w;
  }
#pragma unroll
  for (int ofs = 16; ofs > 0; ofs >>= 1) acc += __shfl_down(acc, ofs, 32);
  if (lane == 0) out[e] = acc + b_us[0];
}

// ---------------------------------------------------------------------------
// Host launcher
// ---------------------------------------------------------------------------
extern "C" void kernel_launch(void* const* d_in, const int* in_sizes, int n_in,
                              void* d_out, int out_size, void* d_ws, size_t ws_size,
                              hipStream_t stream) {
  const float* emb      = (const float*)d_in[0];
  const float* features = (const float*)d_in[1];
  // d_in[2]=W_z, d_in[3]=b_z : provably unused by the outputs.
  const float* W_lin    = (const float*)d_in[4];   // (2H, H) row-major
  const float* b_lin    = (const float*)d_in[5];
  const float* W_us     = (const float*)d_in[6];   // (H, 1) == H floats
  const float* b_us     = (const float*)d_in[7];
  const int*   src1     = (const int*)d_in[8];
  const int*   dst1     = (const int*)d_in[9];
  const int*   src0     = (const int*)d_in[10];
  const int*   dst0     = (const int*)d_in[11];
  float* out = (float*)d_out;

  const int E1 = in_sizes[8];   // 100000
  const int E0 = in_sizes[10];  // 250000

  // Workspace layout: A | B1 | B0 | Asum(->A0) | cnt | Wp_lo | Wp_hi (~94.5 MB)
  float* A    = (float*)d_ws;
  float* B1   = A    + (size_t)GN2 * GH;
  float* B0   = B1   + (size_t)GN1 * GH;
  float* Asum = B0   + (size_t)GN0 * GH;
  float* cnt  = Asum + (size_t)GN1 * GH;
  __bf16* Wp_lo = (__bf16*)(cnt + GN1);           // 32B-aligned (offset % 32 == 0)
  __bf16* Wp_hi = Wp_lo + (size_t)GH * GH;

  const float* W_lo = W_lin;                       // rows 0..H-1   (root half)
  const float* W_hi = W_lin + (size_t)GH * GH;     // rows H..2H-1  (feature half)

  // 1) Zero the accumulation region (Asum + cnt, contiguous).
  zero_kernel<<<2048, 256, 0, stream>>>(Asum, GN1 * GH + GN1);

  // 2) Pre-pack both W halves into bf16 WMMA B-operand order (one shot).
  pack_w_bf16<<<16, 256, 0, stream>>>(W_lo, Wp_lo);
  pack_w_bf16<<<16, 256, 0, stream>>>(W_hi, Wp_hi);

  // 3) Node projections via bf16 WMMA GEMMs.
  proj_gemm_bf16<<<GN2 / 16, 256, 0, stream>>>(emb, Wp_lo, A, GN2);
  proj_gemm_bf16<<<GN1 / 16, 256, 0, stream>>>(features + (size_t)GN2 * GH,
                                               Wp_hi, B1, GN1);
  proj_gemm_bf16<<<GN0 / 16, 256, 0, stream>>>(features + (size_t)(GN2 + GN1) * GH,
                                               Wp_hi, B0, GN0);

  // 4) Segment mean of A over dst1 -> A0 (in Asum).
  scatter_add_kernel<<<(E1 + 3) / 4, 256, 0, stream>>>(A, src1, dst1, Asum, cnt, E1);
  scale_kernel<<<GN1, GH, 0, stream>>>(Asum, cnt);

  // 5) Edge outputs, concatenated [out1 ; out0].
  edge_out_kernel<<<(E1 + 7) / 8, 256, 0, stream>>>(A, B1, src1, dst1,
                                                    b_lin, W_us, b_us, out, E1);
  edge_out_kernel<<<(E0 + 7) / 8, 256, 0, stream>>>(Asum, B0, src0, dst0,
                                                    b_lin, W_us, b_us, out + E1, E0);
}
